// MambaBranch_42554535969384
// MI455X (gfx1250) — matrix operand order repaired and checked by reading, not compiled, not run on hardware
//
#include <hip/hip_runtime.h>
#include <hip/hip_bf16.h>
#include <math.h>

// ---------------------------------------------------------------------------
// Shapes (fixed by the reference)
// ---------------------------------------------------------------------------
#define B_  2
#define D_  2048
#define L_  2048
#define K_  4
#define R_  128
#define N_  16
#define E_  (R_ + 2 * N_)   // 160
#define M_  (B_ * L_)       // 4096

typedef __attribute__((ext_vector_type(16))) _Float16 v16h;
typedef __attribute__((ext_vector_type(8)))  _Float16 v8h;
typedef __attribute__((ext_vector_type(8)))  float    v8f;

// ---------------------------------------------------------------------------
// WMMA fragment loaders (layouts per CDNA5 ISA 7.12.2, wave32)
// A (16x32 f16, row-major source, leading dim ld):
//   lane<16: M=lane,    halves 0..7 = K 0..7,  halves 8..15 = K 16..23
//   lane>=16: M=lane-16, halves 0..7 = K 8..15, halves 8..15 = K 24..31
// ---------------------------------------------------------------------------
__device__ __forceinline__ v16h load_frag_a(const _Float16* __restrict__ A,
                                            int ld, int m0, int k0, int lane) {
    int row = m0 + (lane & 15);
    int kb  = k0 + ((lane >> 4) << 3);
    const _Float16* p = A + (size_t)row * ld + kb;
    v8h lo = *(const v8h*)p;
    v8h hi = *(const v8h*)(p + 16);
    v16h r;
#pragma unroll
    for (int i = 0; i < 8; ++i) { r[i] = lo[i]; r[i + 8] = hi[i]; }
    return r;
}

// B (32x16 f16): Bmat[k][n] = W[n][k], W row-major [Ntotal, Kld].
//   lane<16: N=lane,    halves 0..7 = K 0..7,  halves 8..15 = K 16..23
//   lane>=16: N=lane-16, halves 0..7 = K 8..15, halves 8..15 = K 24..31
__device__ __forceinline__ v16h load_frag_b(const _Float16* __restrict__ W,
                                            int ld, int n0, int k0, int lane) {
    int row = n0 + (lane & 15);
    int kb  = k0 + ((lane >> 4) << 3);
    const _Float16* p = W + (size_t)row * ld + kb;
    v8h lo = *(const v8h*)p;
    v8h hi = *(const v8h*)(p + 16);
    v16h r;
#pragma unroll
    for (int i = 0; i < 8; ++i) { r[i] = lo[i]; r[i + 8] = hi[i]; }
    return r;
}

__device__ __forceinline__ float silu_f(float x) {
    return x / (1.0f + __expf(-x));
}

// ---------------------------------------------------------------------------
// Kernel 1: f32 -> f16 weight conversion (xproj_w, dt_w)
// ---------------------------------------------------------------------------
__global__ void __launch_bounds__(256)
cvt_f32_f16_kernel(const float* __restrict__ src, _Float16* __restrict__ dst, int n) {
    int i = blockIdx.x * blockDim.x + threadIdx.x;
    if (i < n) dst[i] = (_Float16)src[i];
}

// ---------------------------------------------------------------------------
// Kernel 2: causal depthwise conv1d (K=4) + bias + SiLU, with LDS transpose.
// Reads hidden [B,D,L]; writes u as [B,L,D] in f32 (scan) and f16 (WMMA A).
// block (32,8): tile = 32 d x 32 l (with 3-sample halo in l).
// ---------------------------------------------------------------------------
__global__ void __launch_bounds__(256)
conv_silu_kernel(const float* __restrict__ hs, const float* __restrict__ cw,
                 const float* __restrict__ cb, float* __restrict__ u32,
                 _Float16* __restrict__ u16) {
    __shared__ float sh[32][37];   // 37: odd stride -> conflict-free column reads
    const int b  = blockIdx.z;
    const int d0 = blockIdx.y * 32;
    const int l0 = blockIdx.x * 32;
    const int tx = threadIdx.x, ty = threadIdx.y;

#pragma unroll
    for (int i = 0; i < 4; ++i) {
        int dl = ty + 8 * i;
        const float* row = hs + ((size_t)b * D_ + (d0 + dl)) * L_;
        int gl0 = l0 - 3 + tx;
        sh[dl][tx] = (gl0 >= 0) ? row[gl0] : 0.0f;
        int c1 = tx + 32;
        if (c1 < 35) sh[dl][c1] = row[l0 - 3 + c1];  // l0-3+c1 <= l0+31 < L
    }
    __syncthreads();

    const float4 w   = ((const float4*)cw)[d0 + tx];   // conv_w row [4]
    const float bias = cb[d0 + tx];
#pragma unroll
    for (int i = 0; i < 4; ++i) {
        int ll = ty + 8 * i;
        float acc = bias + sh[tx][ll] * w.x + sh[tx][ll + 1] * w.y +
                    sh[tx][ll + 2] * w.z + sh[tx][ll + 3] * w.w;
        float s = silu_f(acc);
        size_t idx = ((size_t)b * L_ + (l0 + ll)) * D_ + d0 + tx;  // [B,L,D]
        u32[idx] = s;
        u16[idx] = (_Float16)s;
    }
}

// ---------------------------------------------------------------------------
// Kernel 3: GEMM1  ssm[M,160] = u16[M,2048] x xw16[160,2048]^T  (WMMA f16->f32)
// Each wave: 4 M-tiles x 1 N-tile (B-fragment reused 4x per k-step).
// 640 waves total -> 80 blocks x 256 threads.
// ---------------------------------------------------------------------------
__global__ void __launch_bounds__(256)
gemm_xproj_kernel(const _Float16* __restrict__ A, const _Float16* __restrict__ W,
                  float* __restrict__ C) {
    const int lane = threadIdx.x & 31;
    const int wave = (blockIdx.x * blockDim.x + threadIdx.x) >> 5;
    const int ntile = wave % (E_ / 16);            // 10 N-tiles
    const int mt4   = wave / (E_ / 16);            // 64 groups of 4 M-tiles
    const int m0 = mt4 * 64, n0 = ntile * 16;

    v8f acc[4] = {v8f{}, v8f{}, v8f{}, v8f{}};
    for (int k0 = 0; k0 < D_; k0 += 32) {
        v16h bf = load_frag_b(W, D_, n0, k0, lane);
#pragma unroll
        for (int i = 0; i < 4; ++i) {
            v16h af = load_frag_a(A, D_, m0 + 16 * i, k0, lane);
            acc[i] = __builtin_amdgcn_wmma_f32_16x16x32_f16(
                false, af, false, bf, (short)0, acc[i], false, false);
        }
    }
    const int n    = n0 + (lane & 15);
    const int mrow = m0 + ((lane >> 4) << 3);
#pragma unroll
    for (int i = 0; i < 4; ++i)
#pragma unroll
        for (int v = 0; v < 8; ++v)
            C[(size_t)(mrow + 16 * i + v) * E_ + n] = acc[i][v];
}

// ---------------------------------------------------------------------------
// Kernel 4: row-wise RMSNorm of ssm -> dt(f16,128), Bn(f32,16), Cn(f32,16).
// One wave per row; 4096 rows -> 512 blocks x 256.
// ---------------------------------------------------------------------------
__global__ void __launch_bounds__(256)
rmsnorm_kernel(const float* __restrict__ ssm, const float* __restrict__ dt_ln,
               const float* __restrict__ b_ln, const float* __restrict__ c_ln,
               _Float16* __restrict__ dtn, float* __restrict__ Bn,
               float* __restrict__ Cn) {
    const int lane = threadIdx.x & 31;
    const int row  = (blockIdx.x * blockDim.x + threadIdx.x) >> 5;
    const float* r = ssm + (size_t)row * E_;

    // dt: 128 elems, 4 per lane
    float v[4];
    float ss = 0.0f;
#pragma unroll
    for (int i = 0; i < 4; ++i) { v[i] = r[lane * 4 + i]; ss += v[i] * v[i]; }
#pragma unroll
    for (int m = 16; m >= 1; m >>= 1) ss += __shfl_xor(ss, m, 32);
    float sc = rsqrtf(ss / (float)R_ + 1e-6f);
#pragma unroll
    for (int i = 0; i < 4; ++i)
        dtn[(size_t)row * R_ + lane * 4 + i] =
            (_Float16)(v[i] * sc * dt_ln[lane * 4 + i]);

    // B (lanes 0-15) and C (lanes 16-31): independent 16-lane reductions
    float x  = r[R_ + lane];
    float s2 = x * x;
#pragma unroll
    for (int m = 8; m >= 1; m >>= 1) s2 += __shfl_xor(s2, m, 32);
    float sc2 = rsqrtf(s2 / (float)N_ + 1e-6f);
    float lnw = (lane < 16) ? b_ln[lane] : c_ln[lane - 16];
    float o   = x * sc2 * lnw;
    if (lane < 16) Bn[(size_t)row * N_ + lane]      = o;
    else           Cn[(size_t)row * N_ + lane - 16] = o;
}

// ---------------------------------------------------------------------------
// Kernel 5: GEMM2  delta[M,2048] = softplus(dtn[M,128] x dtw16[2048,128]^T + b)
// Each wave: 1 M-tile x 4 N-tiles (A-fragment reused 4x). 8192 waves -> 1024 blks.
// ---------------------------------------------------------------------------
__global__ void __launch_bounds__(256)
gemm_delta_kernel(const _Float16* __restrict__ A, const _Float16* __restrict__ W,
                  const float* __restrict__ dt_b, float* __restrict__ delta) {
    const int lane = threadIdx.x & 31;
    const int wave = (blockIdx.x * blockDim.x + threadIdx.x) >> 5;
    const int nt4  = wave % (D_ / 64);             // 32
    const int mt   = wave / (D_ / 64);             // 0..255
    const int m0 = mt * 16, n0 = nt4 * 64;

    v8f acc[4] = {v8f{}, v8f{}, v8f{}, v8f{}};
    for (int k0 = 0; k0 < R_; k0 += 32) {
        v16h af = load_frag_a(A, R_, m0, k0, lane);
#pragma unroll
        for (int j = 0; j < 4; ++j) {
            v16h bf = load_frag_b(W, R_, n0 + 16 * j, k0, lane);
            acc[j] = __builtin_amdgcn_wmma_f32_16x16x32_f16(
                false, af, false, bf, (short)0, acc[j], false, false);
        }
    }
    const int mrow = m0 + ((lane >> 4) << 3);
#pragma unroll
    for (int j = 0; j < 4; ++j) {
        int n = n0 + 16 * j + (lane & 15);
        float bias = dt_b[n];
#pragma unroll
        for (int v = 0; v < 8; ++v) {
            float x  = acc[j][v] + bias;
            float sp = (x > 20.0f) ? x : log1pf(__expf(x));   // softplus
            delta[(size_t)(mrow + v) * D_ + n] = sp;
        }
    }
}

// ---------------------------------------------------------------------------
// Kernel 6: selective scan (serial in L) + skip + gate-SiLU epilogue.
// block = 256 threads = 16 d x 16 n; grid = (D/16, B). 16-lane shfl_xor tree
// for y = sum_n s*C. gate[b,d,:] streams contiguously over the l-loop.
// ---------------------------------------------------------------------------
__global__ void __launch_bounds__(256)
scan_kernel(const float* __restrict__ delta, const float* __restrict__ u,
            const float* __restrict__ Bn, const float* __restrict__ Cn,
            const float* __restrict__ A_log, const float* __restrict__ Dp,
            const float* __restrict__ gate, float* __restrict__ out) {
    const int b  = blockIdx.y;
    const int d0 = blockIdx.x * 16;
    const int n  = threadIdx.x & 15;
    const int dl = threadIdx.x >> 4;
    const int d  = d0 + dl;

    const float a    = -__expf(A_log[d * N_ + n]);
    const float dpar = Dp[d];
    const float* grow = gate + ((size_t)b * D_ + d) * L_;
    float s = 0.0f;

    for (int l = 0; l < L_; ++l) {
        size_t row = (size_t)b * L_ + l;
        float dt = delta[row * D_ + d];
        float uu = u[row * D_ + d];
        float Bt = Bn[row * N_ + n];
        float Ct = Cn[row * N_ + n];
        float dA = __expf(dt * a);
        s = dA * s + (dt * uu) * Bt;
        float y = s * Ct;
        y += __shfl_xor(y, 1, 32);
        y += __shfl_xor(y, 2, 32);
        y += __shfl_xor(y, 4, 32);
        y += __shfl_xor(y, 8, 32);
        if (n == 0) {
            float g = grow[l];
            out[row * D_ + d] = (y + uu * dpar) * silu_f(g);
        }
    }
}

// ---------------------------------------------------------------------------
// Host launch
// ---------------------------------------------------------------------------
extern "C" void kernel_launch(void* const* d_in, const int* in_sizes, int n_in,
                              void* d_out, int out_size, void* d_ws, size_t ws_size,
                              hipStream_t stream) {
    const float* hidden = (const float*)d_in[0];   // [B,D,L]
    const float* gate   = (const float*)d_in[1];   // [B,D,L]
    const float* conv_w = (const float*)d_in[2];   // [D,4]
    const float* conv_b = (const float*)d_in[3];   // [D]
    const float* xprojw = (const float*)d_in[4];   // [160,D]
    const float* dt_w   = (const float*)d_in[5];   // [D,128]
    const float* dt_b   = (const float*)d_in[6];   // [D]
    const float* A_log  = (const float*)d_in[7];   // [D,16]
    const float* Dp     = (const float*)d_in[8];   // [D]
    const float* dt_ln  = (const float*)d_in[9];   // [128]
    const float* b_ln   = (const float*)d_in[10];  // [16]
    const float* c_ln   = (const float*)d_in[11];  // [16]
    float* out = (float*)d_out;                    // [B,L,D]

    // workspace layout (256B aligned)
    uint8_t* ws = (uint8_t*)d_ws;
    size_t off = 0;
    auto take = [&](size_t bytes) -> void* {
        void* p = ws + off;
        off = (off + bytes + 255) & ~(size_t)255;
        return p;
    };
    float*     u32   = (float*)    take((size_t)M_ * D_ * 4);   // 33.5 MB
    _Float16*  u16   = (_Float16*) take((size_t)M_ * D_ * 2);   // 16.8 MB
    float*     dlt   = (float*)    take((size_t)M_ * D_ * 4);   // 33.5 MB
    float*     ssm   = (float*)    take((size_t)M_ * E_ * 4);   //  2.6 MB
    _Float16*  dtn   = (_Float16*) take((size_t)M_ * R_ * 2);   //  1.0 MB
    float*     Bn    = (float*)    take((size_t)M_ * N_ * 4);
    float*     Cn    = (float*)    take((size_t)M_ * N_ * 4);
    _Float16*  xw16  = (_Float16*) take((size_t)E_ * D_ * 2);
    _Float16*  dtw16 = (_Float16*) take((size_t)D_ * R_ * 2);
    (void)ws_size; (void)in_sizes; (void)n_in; (void)out_size;

    // 1) weight conversion
    {
        int n1 = E_ * D_;
        cvt_f32_f16_kernel<<<(n1 + 255) / 256, 256, 0, stream>>>(xprojw, xw16, n1);
        int n2 = D_ * R_;
        cvt_f32_f16_kernel<<<(n2 + 255) / 256, 256, 0, stream>>>(dt_w, dtw16, n2);
    }
    // 2) conv + silu + transpose
    {
        dim3 grid(L_ / 32, D_ / 32, B_), block(32, 8, 1);
        conv_silu_kernel<<<grid, block, 0, stream>>>(hidden, conv_w, conv_b, u32, u16);
    }
    // 3) xproj GEMM (WMMA)
    {
        int waves = (M_ / 64) * (E_ / 16);          // 640
        gemm_xproj_kernel<<<waves * 32 / 256, 256, 0, stream>>>(u16, xw16, ssm);
    }
    // 4) RMSNorm
    {
        rmsnorm_kernel<<<M_ * 32 / 256, 256, 0, stream>>>(ssm, dt_ln, b_ln, c_ln,
                                                          dtn, Bn, Cn);
    }
    // 5) delta GEMM (WMMA) + softplus
    {
        int waves = (M_ / 16) * (D_ / 64);          // 8192
        gemm_delta_kernel<<<waves * 32 / 256, 256, 0, stream>>>(dtn, dtw16, dt_b, dlt);
    }
    // 6) selective scan + gated output
    {
        dim3 grid(D_ / 16, B_, 1);
        scan_kernel<<<grid, 256, 0, stream>>>(dlt, u32, Bn, Cn, A_log, Dp, gate, out);
    }
}
